// MambaTimePredictor_46961172414827
// MI455X (gfx1250) — matrix operand order repaired and checked by reading, not compiled, not run on hardware
//
#include <hip/hip_runtime.h>
#include <hip/hip_bf16.h>
#include <math.h>

// ---------------- model constants (match reference) ----------------
#define B_   2
#define L_   2048
#define M_   (B_ * L_)     // 4096 token rows
#define DM_  256
#define HID_ 256
#define ED_  512
#define NS_  16
#define DTR_ 16
#define NL_  2
#define NC_  3

typedef _Float16 half_t;
typedef half_t v16h __attribute__((ext_vector_type(16)));
typedef half_t v8h  __attribute__((ext_vector_type(8)));
typedef float  v8f  __attribute__((ext_vector_type(8)));

// =====================================================================
// WMMA GEMM:  C[M,N] = act( A[M,K](f16) * W[N,K](f16)^T + bias ) + resid
// Requirements (guaranteed by host code): Mv % 32 == 0, Kv % 32 == 0.
// Each wave computes a 32(M) x 16(N) tile: two A fragments share one B
// fragment -> two v_wmma_f32_16x16x32_f16 per K-step, no branches in the
// inner loop, all loads are contiguous 16B (global_load_b128).
// Fragment layouts per the CDNA5 ISA tables:
//   A (16x32 f16): lane l -> M = l&15, g = l>>4; halves 0..7 = K[8g..8g+8),
//                  halves 8..15 = K[16+8g..16+8g+8)       (contiguous)
//   B (32x16 f16): lane l -> N = l&15, g = l>>4; halves 0..7 = K[8g..8g+8), 8..15 = K[16+8g..+8)
//   C/D (16x16 f32): VGPR r -> row r + 8g, col = lane&15
// act: 0 = none, 1 = softplus
// =====================================================================
__global__ __launch_bounds__(128)
void gemm_wmma_f16(const half_t* __restrict__ A, int lda,
                   const half_t* __restrict__ W, int ldw,
                   float* C, int ldc,
                   int Mv, int Nv, int Kv,
                   const float* __restrict__ bias,
                   const float* resid, int ldr,
                   int act)
{
    const int wave = threadIdx.x >> 5;
    const int tile = blockIdx.x * 4 + wave;
    const int ntn  = (Nv + 15) >> 4;
    const int ntm  = Mv >> 5;                    // 32-row supertiles
    if (tile >= ntm * ntn) return;               // wave-uniform: EXEC stays all-1s
    const int m0   = (tile / ntn) << 5;
    const int n0   = (tile % ntn) << 4;
    const int lane = threadIdx.x & 31;
    const int g    = lane >> 4;
    const int hr   = lane & 15;

    const half_t* __restrict__ arow0 = A + (size_t)(m0 + hr) * lda;
    const half_t* __restrict__ arow1 = arow0 + (size_t)16 * lda;
    const half_t* __restrict__ wrow  = W + (size_t)(n0 + hr) * ldw;

    v8f acc0 = {};
    v8f acc1 = {};
    for (int k0 = 0; k0 < Kv; k0 += 32) {
        // A fragments: halves 0..7 = K[8g..8g+8), halves 8..15 = K[16+8g..+8)
        v8h a0lo = *(const v8h*)(arow0 + k0 + 8 * g);
        v8h a0hi = *(const v8h*)(arow0 + k0 + 16 + 8 * g);
        v8h a1lo = *(const v8h*)(arow1 + k0 + 8 * g);
        v8h a1hi = *(const v8h*)(arow1 + k0 + 16 + 8 * g);
        // B fragment: halves 0..7 = K[8g..8g+8), halves 8..15 = K[16+8g..+8)
        v8h blo  = *(const v8h*)(wrow + k0 + 8 * g);
        v8h bhi  = *(const v8h*)(wrow + k0 + 16 + 8 * g);

        v16h a0 = __builtin_shufflevector(a0lo, a0hi, 0, 1, 2, 3, 4, 5, 6, 7,
                                          8, 9, 10, 11, 12, 13, 14, 15);
        v16h a1 = __builtin_shufflevector(a1lo, a1hi, 0, 1, 2, 3, 4, 5, 6, 7,
                                          8, 9, 10, 11, 12, 13, 14, 15);
        v16h b  = __builtin_shufflevector(blo,  bhi,  0, 1, 2, 3, 4, 5, 6, 7,
                                          8, 9, 10, 11, 12, 13, 14, 15);

        acc0 = __builtin_amdgcn_wmma_f32_16x16x32_f16(
            false, a0, false, b, (short)0, acc0, false, false);
        acc1 = __builtin_amdgcn_wmma_f32_16x16x32_f16(
            false, a1, false, b, (short)0, acc1, false, false);

        if (k0 + 32 < Kv) {                      // gfx1250 global_prefetch path
            __builtin_prefetch(arow0 + k0 + 32, 0, 1);
            __builtin_prefetch(arow1 + k0 + 32, 0, 1);
            __builtin_prefetch(wrow + k0 + 32, 0, 1);
        }
    }

    const int   col = n0 + hr;
    const float bv  = bias ? bias[col] : 0.f;
#pragma unroll
    for (int r = 0; r < 8; ++r) {
        const int row = m0 + r + 8 * g;
        float v = acc0[r] + bv;
        if (act == 1) v = (v > 20.f) ? v : log1pf(__expf(v));   // softplus
        if (resid) v += resid[(size_t)row * (size_t)ldr + col];
        C[(size_t)row * (size_t)ldc + col] = v;
    }
#pragma unroll
    for (int r = 0; r < 8; ++r) {
        const int row = m0 + 16 + r + 8 * g;
        float v = acc1[r] + bv;
        if (act == 1) v = (v > 20.f) ? v : log1pf(__expf(v));
        if (resid) v += resid[(size_t)row * (size_t)ldr + col];
        C[(size_t)row * (size_t)ldc + col] = v;
    }
}

// ---------------- elementwise / support kernels ----------------
__global__ void cast_f2h_kernel(const float* __restrict__ in, half_t* __restrict__ out, int n)
{
    int i = blockIdx.x * blockDim.x + threadIdx.x;
    if (i < n) out[i] = (half_t)in[i];
}

// dst[r*dstride + c] = (c < scols) ? (f16)src[r*sstride + c] : 0
// used to zero-pad the K=16 dt-projection operands up to K=32
__global__ void cast_pad_kernel(const float* __restrict__ src, int sstride, int scols,
                                half_t* __restrict__ dst, int dstride, int rows)
{
    int i = blockIdx.x * blockDim.x + threadIdx.x;
    if (i >= rows * dstride) return;
    int c = i % dstride;
    int r = i / dstride;
    dst[i] = (c < scols) ? (half_t)src[(size_t)r * sstride + c] : (half_t)0.f;
}

__global__ void mfreq_kernel(const float* __restrict__ t, float* __restrict__ th)
{
    int idx = blockIdx.x * blockDim.x + threadIdx.x;   // M_ * 128
    if (idx >= M_ * (DM_ / 2)) return;
    int i   = idx % (DM_ / 2);
    int row = idx / (DM_ / 2);
    float f   = __powf(10000.f, -(float)i / (float)(DM_ / 2));
    float ang = t[row] * f;
    th[(size_t)row * DM_ + i]            = __sinf(ang);
    th[(size_t)row * DM_ + DM_ / 2 + i]  = __cosf(ang);
}

// RMSNorm over DM_, writes f16 for the in_proj GEMM
__global__ void rmsnorm_h_kernel(const float* __restrict__ x, const float* __restrict__ w,
                                 half_t* __restrict__ out)
{
    __shared__ float sh[16];
    const int row = blockIdx.x;
    const float* xr = x + (size_t)row * DM_;
    float v = 0.f;
    for (int i = threadIdx.x; i < DM_; i += blockDim.x) { float t = xr[i]; v += t * t; }
    for (int o = 16; o > 0; o >>= 1) v += __shfl_down(v, o, 32);
    const int wid = threadIdx.x >> 5, ln = threadIdx.x & 31;
    if (ln == 0) sh[wid] = v;
    __syncthreads();
    if (threadIdx.x == 0) {
        float s = 0.f;
        int nw = (blockDim.x + 31) >> 5;
        for (int i = 0; i < nw; ++i) s += sh[i];
        sh[15] = rsqrtf(s / (float)DM_ + 1e-5f);
    }
    __syncthreads();
    const float r = sh[15];
    for (int i = threadIdx.x; i < DM_; i += blockDim.x)
        out[(size_t)row * DM_ + i] = (half_t)(xr[i] * r * w[i]);
}

// causal depthwise conv (K=4) + bias + SiLU; xin = xz[..., :ED]
__global__ void conv_silu_kernel(const float* __restrict__ xz,
                                 const float* __restrict__ cw, const float* __restrict__ cb,
                                 float* __restrict__ xcf, half_t* __restrict__ xch)
{
    int idx = blockIdx.x * blockDim.x + threadIdx.x;   // M_ * ED_
    if (idx >= M_ * ED_) return;
    int e  = idx % ED_;
    int bl = idx / ED_;
    int l  = bl % L_;
    int b  = bl / L_;
    float s = cb[e];
#pragma unroll
    for (int k = 0; k < 4; ++k) {
        int ls = l + k - 3;
        if (ls >= 0) s += cw[e * 4 + k] * xz[((size_t)b * L_ + ls) * (2 * ED_) + e];
    }
    float sv = s / (1.f + __expf(-s));
    xcf[idx] = sv;
    xch[idx] = (half_t)sv;
}

// selective-scan: h_l = exp(dt*A)*h_{l-1} + dt*xin*B ; y = <h,C> + D*xin
// one thread per (b, e, n); reduce over n (16 lanes) via shfl_xor
__global__ void ssm_scan_kernel(const float* __restrict__ dt, const float* __restrict__ xin,
                                const float* __restrict__ Bm, const float* __restrict__ Cm,
                                const float* __restrict__ A_log, const float* __restrict__ Dw,
                                float* __restrict__ y)
{
    int tid = blockIdx.x * blockDim.x + threadIdx.x;   // B_*ED_*NS_ = 16384 exactly
    int n = tid % NS_;
    int e = (tid / NS_) % ED_;
    int b = tid / (NS_ * ED_);
    float Aval = -__expf(A_log[e * NS_ + n]);
    float Dval = Dw[e];
    float h = 0.f;
    for (int l = 0; l < L_; ++l) {
        size_t bl = (size_t)b * L_ + l;
        float dtv = dt[bl * ED_ + e];
        float xv  = xin[bl * ED_ + e];
        float Bv  = Bm[bl * (DTR_ + 2 * NS_) + n];
        float Cv  = Cm[bl * (DTR_ + 2 * NS_) + n];
        h = __expf(dtv * Aval) * h + dtv * xv * Bv;
        float c = h * Cv;
        for (int o = 8; o > 0; o >>= 1) c += __shfl_xor(c, o, 16);
        if (n == 0) y[bl * ED_ + e] = c + Dval * xv;
    }
}

// yz = (f16)( y * silu(z) ),  z = xz[..., ED:]
__global__ void gate_mul_kernel(const float* __restrict__ y, const float* __restrict__ xz,
                                half_t* __restrict__ out)
{
    int idx = blockIdx.x * blockDim.x + threadIdx.x;   // M_ * ED_
    if (idx >= M_ * ED_) return;
    int e = idx % ED_;
    size_t bl = idx / ED_;
    float z = xz[bl * (2 * ED_) + ED_ + e];
    out[idx] = (half_t)(y[idx] * (z / (1.f + __expf(-z))));
}

// time_h += delta_lin * sigmoid(gate_lin)
__global__ void delta_add_kernel(float* __restrict__ th, const float* __restrict__ d,
                                 const float* __restrict__ g)
{
    int i = blockIdx.x * blockDim.x + threadIdx.x;     // M_ * DM_
    if (i < M_ * DM_) th[i] += d[i] * (1.f / (1.f + __expf(-g[i])));
}

// mixture heads: pi = softmax(lin), mu = mod(lin, 2pi), kappa = clip(softplus+1e-4)
__global__ void heads_kernel(const float* __restrict__ th,
                             const float* __restrict__ pw, const float* __restrict__ pb,
                             const float* __restrict__ mw, const float* __restrict__ mb,
                             const float* __restrict__ kw, const float* __restrict__ kb,
                             float* __restrict__ out)
{
    int row = blockIdx.x * blockDim.x + threadIdx.x;
    if (row >= M_) return;
    const float* x = th + (size_t)row * DM_;
    float pv[NC_], mv[NC_], kv[NC_];
    for (int c = 0; c < NC_; ++c) {
        float sp = pb[c], sm = mb[c], sk = kb[c];
        for (int k = 0; k < DM_; ++k) {
            float xv = x[k];
            sp += xv * pw[c * DM_ + k];
            sm += xv * mw[c * DM_ + k];
            sk += xv * kw[c * DM_ + k];
        }
        pv[c] = sp; mv[c] = sm; kv[c] = sk;
    }
    float mx = fmaxf(pv[0], fmaxf(pv[1], pv[2]));
    float e0 = __expf(pv[0] - mx), e1 = __expf(pv[1] - mx), e2 = __expf(pv[2] - mx);
    float inv = 1.f / (e0 + e1 + e2);
    out[row * 3 + 0] = e0 * inv;
    out[row * 3 + 1] = e1 * inv;
    out[row * 3 + 2] = e2 * inv;
    const float TP = 6.283185307179586f;
    for (int c = 0; c < NC_; ++c) {
        float r = fmodf(mv[c], TP);
        if (r < 0.f) r += TP;
        out[M_ * 3 + row * 3 + c] = r;
    }
    for (int c = 0; c < NC_; ++c) {
        float v  = kv[c];
        float sp = ((v > 20.f) ? v : log1pf(__expf(v))) + 1e-4f;
        out[2 * M_ * 3 + row * 3 + c] = fminf(fmaxf(sp, 0.1f), 20.f);
    }
}

// =====================================================================
extern "C" void kernel_launch(void* const* d_in, const int* in_sizes, int n_in,
                              void* d_out, int out_size, void* d_ws, size_t ws_size,
                              hipStream_t stream)
{
    (void)in_sizes; (void)n_in; (void)out_size;

    // ---- inputs (setup_inputs dict order, nested dicts in insertion order) ----
    const float* poi_emb   = (const float*)d_in[0];
    const float* time_seqs = (const float*)d_in[1];
    struct Tower {
        const float *norm_w, *in_proj, *conv_w, *conv_b, *x_proj,
                    *dt_w, *dt_b, *A_log, *D, *out_proj;
    } tw[2];
    for (int t = 0; t < 2; ++t) {
        int b = 2 + t * 10;
        tw[t].norm_w   = (const float*)d_in[b + 0];
        tw[t].in_proj  = (const float*)d_in[b + 1];
        tw[t].conv_w   = (const float*)d_in[b + 2];
        tw[t].conv_b   = (const float*)d_in[b + 3];
        tw[t].x_proj   = (const float*)d_in[b + 4];
        tw[t].dt_w     = (const float*)d_in[b + 5];
        tw[t].dt_b     = (const float*)d_in[b + 6];
        tw[t].A_log    = (const float*)d_in[b + 7];
        tw[t].D        = (const float*)d_in[b + 8];
        tw[t].out_proj = (const float*)d_in[b + 9];
    }
    const int pbase = 22;
    const float* poi_in_w = (const float*)d_in[pbase + 0];
    const float* poi_in_b = (const float*)d_in[pbase + 1];
    const float* delta_w  = (const float*)d_in[pbase + 2];
    const float* delta_b  = (const float*)d_in[pbase + 3];
    const float* gate_w   = (const float*)d_in[pbase + 4];
    const float* gate_b   = (const float*)d_in[pbase + 5];
    const float* t2p_w    = (const float*)d_in[pbase + 6];
    const float* t2p_b    = (const float*)d_in[pbase + 7];
    const float* pi_w     = (const float*)d_in[pbase + 8];
    const float* pi_b     = (const float*)d_in[pbase + 9];
    const float* mu_w     = (const float*)d_in[pbase + 10];
    const float* mu_b     = (const float*)d_in[pbase + 11];
    const float* kappa_w  = (const float*)d_in[pbase + 12];
    const float* kappa_b  = (const float*)d_in[pbase + 13];

    // ---- workspace bump allocator (256B aligned, recomputed each call) ----
    char*  base = (char*)d_ws;
    size_t off  = 0;
    auto alloc = [&](size_t bytes) -> void* {
        void* p = base + off;
        off = (off + bytes + 255) & ~(size_t)255;
        return p;
    };
    // f32 activations
    float* time_h    = (float*)alloc((size_t)M_ * DM_ * 4);
    float* poi_resid = (float*)alloc((size_t)M_ * DM_ * 4);
    float* poi_h     = (float*)alloc((size_t)M_ * DM_ * 4);
    float* xz        = (float*)alloc((size_t)M_ * 2 * ED_ * 4);
    float* xc_f      = (float*)alloc((size_t)M_ * ED_ * 4);
    float* dbc_f     = (float*)alloc((size_t)M_ * (DTR_ + 2 * NS_) * 4);
    float* dt_f      = (float*)alloc((size_t)M_ * ED_ * 4);
    float* y_f       = (float*)alloc((size_t)M_ * ED_ * 4);
    float* dbuf      = (float*)alloc((size_t)M_ * DM_ * 4);
    float* gbuf      = (float*)alloc((size_t)M_ * DM_ * 4);
    // f16 activations
    half_t* xn_h      = (half_t*)alloc((size_t)M_ * DM_ * 2);
    half_t* xc_h      = (half_t*)alloc((size_t)M_ * ED_ * 2);
    half_t* dbc_pad_h = (half_t*)alloc((size_t)M_ * 32 * 2);      // dt A, K padded to 32
    half_t* yz_h      = (half_t*)alloc((size_t)M_ * ED_ * 2);
    half_t* act_h     = (half_t*)alloc((size_t)M_ * DM_ * 2);
    // f16 weights
    half_t* inproj_h[2], *xproj_h[2], *dtw_pad_h[2], *outproj_h[2];
    for (int t = 0; t < 2; ++t) {
        inproj_h[t]  = (half_t*)alloc((size_t)NL_ * 2 * ED_ * DM_ * 2);
        xproj_h[t]   = (half_t*)alloc((size_t)NL_ * (DTR_ + 2 * NS_) * ED_ * 2);
        dtw_pad_h[t] = (half_t*)alloc((size_t)NL_ * ED_ * 32 * 2); // K padded to 32
        outproj_h[t] = (half_t*)alloc((size_t)NL_ * DM_ * ED_ * 2);
    }
    half_t* poi_in_w_h = (half_t*)alloc((size_t)DM_ * HID_ * 2);
    half_t* delta_w_h  = (half_t*)alloc((size_t)DM_ * DM_ * 2);
    half_t* gate_w_h   = (half_t*)alloc((size_t)DM_ * DM_ * 2);
    half_t* t2p_w_h    = (half_t*)alloc((size_t)HID_ * DM_ * 2);
    if (off > ws_size) return;   // deterministic bail-out if scratch too small

    auto cast = [&](const float* s, half_t* d, int n) {
        cast_f2h_kernel<<<(n + 255) / 256, 256, 0, stream>>>(s, d, n);
    };
    auto cast_pad = [&](const float* s, int sstride, int scols, half_t* d, int dstride, int rows) {
        int n = rows * dstride;
        cast_pad_kernel<<<(n + 255) / 256, 256, 0, stream>>>(s, sstride, scols, d, dstride, rows);
    };
    auto gemm = [&](const half_t* A, int lda, const half_t* W, int ldw,
                    float* C, int ldc, int Mv, int Nv, int Kv,
                    const float* bias, const float* resid, int ldr, int act) {
        int tiles  = (Mv / 32) * ((Nv + 15) / 16);
        int blocks = (tiles + 3) / 4;
        gemm_wmma_f16<<<blocks, 128, 0, stream>>>(A, lda, W, ldw, C, ldc,
                                                  Mv, Nv, Kv, bias, resid, ldr, act);
    };

    // ---- one-time (per launch) weight conversion to f16 ----
    for (int t = 0; t < 2; ++t) {
        cast(tw[t].in_proj,  inproj_h[t],  NL_ * 2 * ED_ * DM_);
        cast(tw[t].x_proj,   xproj_h[t],   NL_ * (DTR_ + 2 * NS_) * ED_);
        cast_pad(tw[t].dt_w, DTR_, DTR_, dtw_pad_h[t], 32, NL_ * ED_);
        cast(tw[t].out_proj, outproj_h[t], NL_ * DM_ * ED_);
    }
    cast(poi_in_w, poi_in_w_h, DM_ * HID_);
    cast(delta_w,  delta_w_h,  DM_ * DM_);
    cast(gate_w,   gate_w_h,   DM_ * DM_);
    cast(t2p_w,    t2p_w_h,    HID_ * DM_);

    // ---- mamba block as a host-side sequence ----
    auto mamba = [&](float* x, const Tower& T, const half_t* ipj, const half_t* xpj,
                     const half_t* dwh, const half_t* opj, int l) {
        rmsnorm_h_kernel<<<M_, 256, 0, stream>>>(x, T.norm_w + l * DM_, xn_h);
        gemm(xn_h, DM_, ipj + (size_t)l * 2 * ED_ * DM_, DM_,
             xz, 2 * ED_, M_, 2 * ED_, DM_, nullptr, nullptr, 0, 0);
        conv_silu_kernel<<<(M_ * ED_ + 255) / 256, 256, 0, stream>>>(
            xz, T.conv_w + (size_t)l * ED_ * 4, T.conv_b + l * ED_, xc_f, xc_h);
        gemm(xc_h, ED_, xpj + (size_t)l * (DTR_ + 2 * NS_) * ED_, ED_,
             dbc_f, DTR_ + 2 * NS_, M_, DTR_ + 2 * NS_, ED_, nullptr, nullptr, 0, 0);
        // pad dt input (first DTR_ cols of dbc) to K=32
        cast_pad(dbc_f, DTR_ + 2 * NS_, DTR_, dbc_pad_h, 32, M_);
        gemm(dbc_pad_h, 32, dwh + (size_t)l * ED_ * 32, 32,
             dt_f, ED_, M_, ED_, 32, T.dt_b + l * ED_, nullptr, 0, /*softplus*/1);
        ssm_scan_kernel<<<(B_ * ED_ * NS_) / 256, 256, 0, stream>>>(
            dt_f, xc_f, dbc_f + DTR_, dbc_f + DTR_ + NS_,
            T.A_log + (size_t)l * ED_ * NS_, T.D + l * ED_, y_f);
        gate_mul_kernel<<<(M_ * ED_ + 255) / 256, 256, 0, stream>>>(y_f, xz, yz_h);
        gemm(yz_h, ED_, opj + (size_t)l * DM_ * ED_, ED_,
             x, DM_, M_, DM_, ED_, nullptr, /*resid=*/x, DM_, 0);   // += input x
    };

    // ---- forward pass ----
    mfreq_kernel<<<(M_ * (DM_ / 2) + 255) / 256, 256, 0, stream>>>(time_seqs, time_h);
    hipMemcpyAsync(poi_resid, poi_emb, (size_t)M_ * DM_ * 4, hipMemcpyDeviceToDevice, stream);
    cast(poi_resid, act_h, M_ * DM_);
    gemm(act_h, HID_, poi_in_w_h, HID_, poi_h, DM_, M_, DM_, HID_, poi_in_b, nullptr, 0, 0);

    for (int l = 0; l < NL_; ++l) {
        mamba(poi_h, tw[0], inproj_h[0], xproj_h[0], dtw_pad_h[0], outproj_h[0], l);
        cast(poi_h, act_h, M_ * DM_);
        gemm(act_h, DM_, delta_w_h, DM_, dbuf, DM_, M_, DM_, DM_, delta_b, nullptr, 0, 0);
        gemm(act_h, DM_, gate_w_h,  DM_, gbuf, DM_, M_, DM_, DM_, gate_b,  nullptr, 0, 0);
        mamba(time_h, tw[1], inproj_h[1], xproj_h[1], dtw_pad_h[1], outproj_h[1], l);
        delta_add_kernel<<<(M_ * DM_ + 255) / 256, 256, 0, stream>>>(time_h, dbuf, gbuf);
        cast(time_h, act_h, M_ * DM_);
        gemm(act_h, DM_, t2p_w_h, DM_, poi_resid, DM_, M_, HID_, DM_,
             t2p_b, /*resid=*/poi_resid, DM_, 0);                   // residual +=
        if (l + 1 < NL_) {
            cast(poi_resid, act_h, M_ * DM_);
            gemm(act_h, HID_, poi_in_w_h, HID_, poi_h, DM_, M_, DM_, HID_,
                 poi_in_b, nullptr, 0, 0);
        }
    }

    // ---- heads + output assembly: [pi | mu | kappa | poi_residual] ----
    float* out = (float*)d_out;
    heads_kernel<<<(M_ + 255) / 256, 256, 0, stream>>>(
        time_h, pi_w, pi_b, mu_w, mu_b, kappa_w, kappa_b, out);
    hipMemcpyAsync(out + 3 * M_ * NC_, poi_resid, (size_t)M_ * DM_ * 4,
                   hipMemcpyDeviceToDevice, stream);
}
